// graph_transf_block4_17497696764593
// MI455X (gfx1250) — matrix operand order repaired and checked by reading, not compiled, not run on hardware
//
#include <hip/hip_runtime.h>
#include <hip/hip_bf16.h>
#include <math.h>
#include <stdint.h>

#define NNODES 10000
#define NEDGES 160000

typedef __attribute__((ext_vector_type(16))) __bf16        v16bf;
typedef __attribute__((ext_vector_type(8)))  float         v8f;
typedef __attribute__((ext_vector_type(4)))  unsigned int  u32x4;

// global (address_space 1) typed pointers: keep GEMM loads on the
// global_load_b128 path (flat loads would tick DScnt and the LDS pipe).
typedef __attribute__((address_space(1))) const u32x4          GCU4;
typedef __attribute__((address_space(1))) const unsigned short GCU16;

union FragU { v16bf v; u32x4 u[2]; };

__device__ __forceinline__ unsigned short f2bf(float f) {
  unsigned u = __float_as_uint(f);
  u += 0x7FFFu + ((u >> 16) & 1u);          // round-to-nearest-even
  return (unsigned short)(u >> 16);
}
__device__ __forceinline__ float bflo(unsigned u) { return __uint_as_float(u << 16); }
__device__ __forceinline__ float bfhi(unsigned u) { return __uint_as_float(u & 0xFFFF0000u); }
// monotonic float<->uint mapping for atomicMax-based segment max
__device__ __forceinline__ unsigned fenc(float f) {
  unsigned u = __float_as_uint(f);
  return (u & 0x80000000u) ? ~u : (u | 0x80000000u);
}
__device__ __forceinline__ float fdec(unsigned u) {
  unsigned v = (u & 0x80000000u) ? (u & 0x7FFFFFFFu) : ~u;
  return __uint_as_float(v);
}

// ---------------------------------------------------------------------------
// Activation pack: f32 [M][K] (optionally ELU) -> bf16 packed [K/32][M][32]
// ---------------------------------------------------------------------------
__global__ __launch_bounds__(256)
void pack_act(const float* __restrict__ src, unsigned short* __restrict__ dst,
              int M, int K, int doElu) {
  size_t i = (size_t)blockIdx.x * 256 + threadIdx.x;
  if (i >= (size_t)M * K) return;
  int k = (int)(i % K);
  int m = (int)(i / K);
  float v = src[i];
  if (doElu) v = (v > 0.f) ? v : (__expf(v) - 1.f);
  dst[((size_t)(k >> 5) * M + m) * 32 + (k & 31)] = f2bf(v);
}

// ---------------------------------------------------------------------------
// Weight pack: f32 W [K][Nout] -> bf16 packed [K/32][Nout][32]
// ---------------------------------------------------------------------------
__global__ __launch_bounds__(256)
void pack_w(const float* __restrict__ W, unsigned short* __restrict__ Wp,
            int K, int Nout) {
  size_t i = (size_t)blockIdx.x * 256 + threadIdx.x;
  if (i >= (size_t)K * Nout) return;
  int kk = (int)(i & 31);
  size_t t = i >> 5;
  int n  = (int)(t % Nout);
  int kb = (int)(t / Nout);
  Wp[i] = f2bf(W[(size_t)(kb * 32 + kk) * Nout + n]);
}

// ---------------------------------------------------------------------------
// WMMA bf16 GEMM: Out[M][Nout] = A[M][K] * B[K][Nout] + bias
// A, B in packed [K/32][row][32] bf16 layout. 128x128 block tile, 4 waves
// (2x2), 64x64 per-wave register tile = 4x4 fragments of
// v_wmma_f32_16x16x32_bf16 (16 wmmas per 32-K chunk, 32 flop/byte from L2).
// Branch-free inner loop; fragment row bases clamped to M-16 up front
// (M % 16 == 0 so clamped loads stay in-bounds); stores are guarded.
// All fragment loads go through address_space(1) pointers -> global_load_b128.
// ---------------------------------------------------------------------------
template <bool OUTBF16>
__global__ __launch_bounds__(128)
void gemm_bf16_wmma(const unsigned short* __restrict__ Ap,
                    const unsigned short* __restrict__ Bp,
                    const float* __restrict__ bias,
                    void* __restrict__ Out,
                    int M, int K, int Nout) {
  const int tid  = threadIdx.x;
  const int wv   = tid >> 5;
  const int lane = tid & 31;
  const int wm = wv >> 1, wn = wv & 1;        // 2 x 2 wave grid
  const int m0 = blockIdx.x * 128 + wm * 64;
  const int n0 = blockIdx.y * 128 + wn * 64;
  const int l16 = lane & 15;
  const int h   = lane >> 4;                  // half-wave selects K-half

  v8f acc[4][4];
  #pragma unroll
  for (int i = 0; i < 4; ++i)
    #pragma unroll
    for (int j = 0; j < 4; ++j)
      #pragma unroll
      for (int r = 0; r < 8; ++r) acc[i][j][r] = 0.f;

  const size_t strideA = (size_t)M * 32;      // bf16 elements per K-block
  const size_t strideB = (size_t)Nout * 32;

  const GCU16* aPtr[4];
  const GCU16* bPtr[4];
  #pragma unroll
  for (int fm = 0; fm < 4; ++fm) {
    int mf = m0 + fm * 16;
    if (mf > M - 16) mf = M - 16;             // clamp: loads stay in-bounds
    aPtr[fm] = (const GCU16*)(uintptr_t)(Ap + (size_t)(mf + l16) * 32 + h * 8);
  }
  #pragma unroll
  for (int fn = 0; fn < 4; ++fn)
    bPtr[fn] = (const GCU16*)(uintptr_t)(Bp + (size_t)(n0 + fn * 16 + l16) * 32 + h * 8);

  const int KB = K >> 5;
  for (int kb = 0; kb < KB; ++kb) {
    FragU a[4], b[4];
    #pragma unroll
    for (int fm = 0; fm < 4; ++fm) {
      a[fm].u[0] = *(const GCU4*)(aPtr[fm]);        // K = h*8    .. h*8+7
      a[fm].u[1] = *(const GCU4*)(aPtr[fm] + 16);   // K = 16+h*8 ..
    }
    #pragma unroll
    for (int fn = 0; fn < 4; ++fn) {
      b[fn].u[0] = *(const GCU4*)(bPtr[fn]);        // K = h*8    .. h*8+7
      b[fn].u[1] = *(const GCU4*)(bPtr[fn] + 16);   // K = 16+h*8 ..
    }
    #pragma unroll
    for (int fm = 0; fm < 4; ++fm)
      #pragma unroll
      for (int fn = 0; fn < 4; ++fn)
        acc[fm][fn] = __builtin_amdgcn_wmma_f32_16x16x32_bf16(
            false, a[fm].v, false, b[fn].v, (short)0, acc[fm][fn], false, false);
    #pragma unroll
    for (int fm = 0; fm < 4; ++fm) aPtr[fm] += strideA;
    #pragma unroll
    for (int fn = 0; fn < 4; ++fn) bPtr[fn] += strideB;
  }

  #pragma unroll
  for (int fn = 0; fn < 4; ++fn) {
    int col = n0 + fn * 16 + l16;
    float bv = bias[col];
    #pragma unroll
    for (int fm = 0; fm < 4; ++fm) {
      int mf = m0 + fm * 16;
      if (mf >= M) continue;                  // uniform store guard
      #pragma unroll
      for (int r = 0; r < 8; ++r) {
        int row = mf + h * 8 + r;             // C/D layout: vgpr r -> M row
        float val = acc[fm][fn][r] + bv;
        if (OUTBF16)
          ((unsigned short*)Out)[(size_t)row * Nout + col] = f2bf(val);
        else
          ((float*)Out)[(size_t)row * Nout + col] = val;
      }
    }
  }
}

// ---------------------------------------------------------------------------
// Per-node init: segment max (encoded -inf) and segment sum (0)
// ---------------------------------------------------------------------------
__global__ __launch_bounds__(256)
void init_nodes(unsigned* __restrict__ mEnc, float* __restrict__ ssum, int n) {
  int i = blockIdx.x * 256 + threadIdx.x;
  if (i < n) { mEnc[i] = fenc(-INFINITY); ssum[i] = 0.f; }
}

// ---------------------------------------------------------------------------
// Pass 1: logits[e] = dot(q[dst], k[src]) / sqrt(C); atomic segment max
// One wave per edge; q,k are bf16 [N][C] (L2 resident).
// ---------------------------------------------------------------------------
__global__ __launch_bounds__(256)
void edge_logits(const unsigned short* __restrict__ q,
                 const unsigned short* __restrict__ k,
                 const long long* __restrict__ src,
                 const long long* __restrict__ dst,
                 float* __restrict__ logits, unsigned* __restrict__ mEnc,
                 int E, int C, float scale) {
  int e = blockIdx.x * 8 + (threadIdx.x >> 5);
  if (e >= E) return;
  int lane = threadIdx.x & 31;
  long long s = src[e], d = dst[e];
  const unsigned short* qp = q + (size_t)d * C;
  const unsigned short* kp = k + (size_t)s * C;
  float sum = 0.f;
  for (int c = lane * 8; c < C; c += 256) {
    u32x4 qa = *(const u32x4*)(qp + c);
    u32x4 ka = *(const u32x4*)(kp + c);
    sum += bflo(qa.x) * bflo(ka.x) + bfhi(qa.x) * bfhi(ka.x);
    sum += bflo(qa.y) * bflo(ka.y) + bfhi(qa.y) * bfhi(ka.y);
    sum += bflo(qa.z) * bflo(ka.z) + bfhi(qa.z) * bfhi(ka.z);
    sum += bflo(qa.w) * bflo(ka.w) + bfhi(qa.w) * bfhi(ka.w);
  }
  #pragma unroll
  for (int off = 16; off > 0; off >>= 1) sum += __shfl_xor(sum, off, 32);
  if (lane == 0) {
    float l = sum * scale;
    logits[e] = l;
    atomicMax(&mEnc[(size_t)d], fenc(l));
  }
}

// ---------------------------------------------------------------------------
// Pass 2: e = exp(logit - m[dst]); atomic segment sum. Thread per edge.
// ---------------------------------------------------------------------------
__global__ __launch_bounds__(256)
void edge_exp(const long long* __restrict__ dst,
              const unsigned* __restrict__ mEnc,
              float* __restrict__ logits, float* __restrict__ ssum, int E) {
  int e = blockIdx.x * 256 + threadIdx.x;
  if (e >= E) return;
  long long d = dst[e];
  float m = fdec(mEnc[(size_t)d]);
  if (!isfinite(m)) m = 0.f;
  float w = __expf(logits[e] - m);
  logits[e] = w;
  atomicAdd(&ssum[(size_t)d], w);
}

// ---------------------------------------------------------------------------
// Pass 3: acc[dst] += v[src] * alpha. One wave per edge; f32 atomics (L2).
// ---------------------------------------------------------------------------
__global__ __launch_bounds__(256)
void edge_scatter(const unsigned short* __restrict__ v,
                  const long long* __restrict__ src,
                  const long long* __restrict__ dst,
                  const float* __restrict__ logits,
                  const float* __restrict__ ssum,
                  float* __restrict__ acc, int E, int C) {
  int e = blockIdx.x * 8 + (threadIdx.x >> 5);
  if (e >= E) return;
  int lane = threadIdx.x & 31;
  long long s = src[e], d = dst[e];
  float alpha = logits[e] / (ssum[(size_t)d] + 1e-16f);
  const unsigned short* vp = v + (size_t)s * C;
  float* ap = acc + (size_t)d * C;
  for (int c = lane * 8; c < C; c += 256) {
    u32x4 va = *(const u32x4*)(vp + c);
    atomicAdd(&ap[c + 0], bflo(va.x) * alpha);
    atomicAdd(&ap[c + 1], bfhi(va.x) * alpha);
    atomicAdd(&ap[c + 2], bflo(va.y) * alpha);
    atomicAdd(&ap[c + 3], bfhi(va.y) * alpha);
    atomicAdd(&ap[c + 4], bflo(va.z) * alpha);
    atomicAdd(&ap[c + 5], bfhi(va.z) * alpha);
    atomicAdd(&ap[c + 6], bflo(va.w) * alpha);
    atomicAdd(&ap[c + 7], bfhi(va.w) * alpha);
  }
}

// ---------------------------------------------------------------------------
// Host orchestration
// ---------------------------------------------------------------------------
static const int FI[4] = {2048, 1024, 2048, 1024};
static const int FO[4] = {1024, 2048, 1024, 2048};

extern "C" void kernel_launch(void* const* d_in, const int* in_sizes, int n_in,
                              void* d_out, int out_size, void* d_ws, size_t ws_size,
                              hipStream_t stream) {
  const int N = NNODES, E = NEDGES;
  const float* x = (const float*)d_in[0];
  const long long* ei = (const long long*)d_in[1];
  const long long* srcI = ei;          // edge_index[0]
  const long long* dstI = ei + E;      // edge_index[1]

  char* wsp = (char*)d_ws;
  size_t off = 0;
  auto wsalloc = [&](size_t bytes) -> void* {
    void* p = wsp + off;
    off += (bytes + 255) & ~(size_t)255;
    return p;
  };
  unsigned short* xpack = (unsigned short*)wsalloc((size_t)N * 2048 * 2);
  unsigned short* qbuf  = (unsigned short*)wsalloc((size_t)N * 2048 * 2);
  unsigned short* kbuf  = (unsigned short*)wsalloc((size_t)N * 2048 * 2);
  unsigned short* vbuf  = (unsigned short*)wsalloc((size_t)N * 2048 * 2);
  float*          accb  = (float*)wsalloc((size_t)N * 2048 * 4);
  unsigned short* wpq   = (unsigned short*)wsalloc((size_t)2048 * 1024 * 2);
  unsigned short* wpk   = (unsigned short*)wsalloc((size_t)2048 * 1024 * 2);
  unsigned short* wpv   = (unsigned short*)wsalloc((size_t)2048 * 1024 * 2);
  unsigned short* wps   = (unsigned short*)wsalloc((size_t)2048 * 1024 * 2);
  float*    logits = (float*)wsalloc((size_t)E * 4);
  unsigned* mEnc   = (unsigned*)wsalloc((size_t)N * 4);
  float*    ssum   = (float*)wsalloc((size_t)N * 4);

  for (int li = 0; li < 4; ++li) {
    const int fi = FI[li], fo = FO[li];
    // --- pack activation (ELU fused from previous layer where needed) ---
    {
      const float* srcAct = (li == 0) ? x : accb;
      int doElu = (li == 1 || li == 3) ? 1 : 0;
      size_t tot = (size_t)N * fi;
      pack_act<<<dim3((unsigned)((tot + 255) / 256)), 256, 0, stream>>>(
          srcAct, xpack, N, fi, doElu);
    }
    // --- pack weights (f32 -> bf16, WMMA-friendly layout) ---
    const float* Wq = (const float*)d_in[2 + li * 8 + 0];
    const float* bq = (const float*)d_in[2 + li * 8 + 1];
    const float* Wk = (const float*)d_in[2 + li * 8 + 2];
    const float* bk = (const float*)d_in[2 + li * 8 + 3];
    const float* Wv = (const float*)d_in[2 + li * 8 + 4];
    const float* bv = (const float*)d_in[2 + li * 8 + 5];
    const float* Ws = (const float*)d_in[2 + li * 8 + 6];
    const float* bs = (const float*)d_in[2 + li * 8 + 7];
    size_t wtot = (size_t)fi * fo;
    unsigned wblk = (unsigned)((wtot + 255) / 256);
    pack_w<<<dim3(wblk), 256, 0, stream>>>(Wq, wpq, fi, fo);
    pack_w<<<dim3(wblk), 256, 0, stream>>>(Wk, wpk, fi, fo);
    pack_w<<<dim3(wblk), 256, 0, stream>>>(Wv, wpv, fi, fo);
    pack_w<<<dim3(wblk), 256, 0, stream>>>(Ws, wps, fi, fo);

    // --- GEMMs: q,k,v (bf16 out) and skip (f32 out -> acc init) ---
    dim3 gg((N + 127) / 128, fo / 128);
    gemm_bf16_wmma<true><<<gg, 128, 0, stream>>>(xpack, wpq, bq, qbuf, N, fi, fo);
    gemm_bf16_wmma<true><<<gg, 128, 0, stream>>>(xpack, wpk, bk, kbuf, N, fi, fo);
    gemm_bf16_wmma<true><<<gg, 128, 0, stream>>>(xpack, wpv, bv, vbuf, N, fi, fo);
    float* accT = (li == 3) ? (float*)d_out : accb;
    gemm_bf16_wmma<false><<<gg, 128, 0, stream>>>(xpack, wps, bs, accT, N, fi, fo);

    // --- edge softmax-attention aggregation ---
    init_nodes<<<dim3((N + 255) / 256), 256, 0, stream>>>(mEnc, ssum, N);
    float scale = 1.0f / sqrtf((float)fo);
    edge_logits<<<dim3((E + 7) / 8), 256, 0, stream>>>(qbuf, kbuf, srcI, dstI,
                                                       logits, mEnc, E, fo, scale);
    edge_exp<<<dim3((E + 255) / 256), 256, 0, stream>>>(dstI, mEnc, logits, ssum, E);
    edge_scatter<<<dim3((E + 7) / 8), 256, 0, stream>>>(vbuf, srcI, dstI, logits,
                                                        ssum, accT, E, fo);
  }
}
